// StackedMamba_12661563588756
// MI455X (gfx1250) — matrix operand order repaired and checked by reading, not compiled, not run on hardware
//
#include <hip/hip_runtime.h>
#include <hip/hip_bf16.h>
#include <cstdint>
#include <cstddef>

#define D_MODEL   768
#define D_INNER   1536
#define D_STATE   128
#define NHEADS    24
#define HEADDIM   64
#define CONV_DIM  1792            // D_INNER + 2*D_STATE
#define D_IN_PROJ 3352            // 2*D_INNER + 2*D_STATE + NHEADS
#define LSEQ      512
#define BATCH     2
#define ROWS      (BATCH*LSEQ)    // 1024
#define LABELS    1000

typedef __attribute__((ext_vector_type(16))) __bf16          v16bf;
typedef __attribute__((ext_vector_type(8)))  float           v8f;
typedef __attribute__((ext_vector_type(8)))  unsigned short  ushort8;

union Frag { ushort8 h[2]; v16bf v; };

__device__ __forceinline__ unsigned short f2bf(float f) {
  unsigned int u = __float_as_uint(f);
  u += 0x7FFFu + ((u >> 16) & 1u);      // round-to-nearest-even
  return (unsigned short)(u >> 16);
}
__device__ __forceinline__ float sigmoid_(float x) { return 1.f / (1.f + expf(-x)); }

__device__ __forceinline__ v8f wmma_bf16(v16bf a, v16bf b, v8f c) {
  return __builtin_amdgcn_wmma_f32_16x16x32_bf16(false, a, false, b, (short)0, c, false, false);
}

// ---------------- embedding gather: x[r][d] = emb[tokens[r]][d] ----------------
__global__ void k_embed(const int* __restrict__ tok, const float* __restrict__ emb,
                        float* __restrict__ x) {
  int e = blockIdx.x * 256 + threadIdx.x;            // over ROWS*D_MODEL
  int r = e / D_MODEL, d = e - r * D_MODEL;
  x[e] = emb[(size_t)tok[r] * D_MODEL + d];
}

// ---------------- layernorm row -> bf16 ----------------
__global__ void k_layernorm(const float* __restrict__ x, const float* __restrict__ w,
                            const float* __restrict__ b, unsigned short* __restrict__ nx) {
  int r = blockIdx.x, t = threadIdx.x;               // 256 threads, 3 elems each
  __shared__ float s1[256], s2[256];
  float v[3]; float s = 0.f, sq = 0.f;
  #pragma unroll
  for (int i = 0; i < 3; ++i) {
    int d = t + i * 256;
    float f = x[(size_t)r * D_MODEL + d];
    v[i] = f; s += f; sq += f * f;
  }
  s1[t] = s; s2[t] = sq; __syncthreads();
  for (int st = 128; st > 0; st >>= 1) { if (t < st) { s1[t] += s1[t+st]; s2[t] += s2[t+st]; } __syncthreads(); }
  float mu = s1[0] * (1.f / D_MODEL);
  float var = s2[0] * (1.f / D_MODEL) - mu * mu;
  float rs = rsqrtf(var + 1e-5f);
  #pragma unroll
  for (int i = 0; i < 3; ++i) {
    int d = t + i * 256;
    nx[(size_t)r * D_MODEL + d] = f2bf((v[i] - mu) * rs * w[d] + b[d]);
  }
}

// ---------------- f32 -> bf16 convert ----------------
__global__ void k_convert(const float* __restrict__ src, unsigned short* __restrict__ dst) {
  int e = blockIdx.x * 256 + threadIdx.x;
  dst[e] = f2bf(src[e]);
}

// ---------------- pack f32 KxN weight into WMMA B-fragment order (bf16) ----------------
// out[(kt*NT + nt)*512 + lane*16 + e]; h = lane/16, col = nt*16+(lane&15)
//   e<8: K = kt*32+8h+e;  e>=8: K = kt*32+16+8h+(e-8)
__global__ void k_pack(const float* __restrict__ W, unsigned short* __restrict__ out,
                       int K, int N, int NT) {
  size_t idx = (size_t)blockIdx.x * 256 + threadIdx.x;   // total = (K/32)*NT*512
  int e = idx & 15; int lane = (idx >> 4) & 31; size_t tile = idx >> 9;
  int nt = (int)(tile % NT); int kt = (int)(tile / NT);
  int col = nt * 16 + (lane & 15);
  int krow = kt * 32 + ((e & 8) << 1) + ((lane >> 4) << 3) + (e & 7);
  float v = (col < N) ? W[(size_t)krow * N + col] : 0.f;
  out[idx] = f2bf(v);
}

// ---------------- bf16 WMMA GEMM: C[M,N] (+bias)(+residual) = A[M,K] @ Bpacked ----------------
// one wave / block (32 threads); wave computes 32x64 output (2 M-tiles x 4 N-tiles)
__global__ void __launch_bounds__(32)
k_gemm_bf16(const unsigned short* __restrict__ A,
            const unsigned short* __restrict__ Bp,
            float* __restrict__ C,
            const float* __restrict__ bias,
            int K, int N, int NT, int addResidual) {
  int lane = threadIdx.x;
  int m0 = blockIdx.x * 32;
  int nt0 = blockIdx.y * 4;
  v8f acc[2][4] = {};
  int row = lane & 15;
  int koff = (lane < 16) ? 0 : 8;
  int KT = K >> 5;
  for (int kt = 0; kt < KT; ++kt) {
    int k0 = kt << 5;
    Frag fa0, fa1;
    const unsigned short* ap0 = A + (size_t)(m0 + row) * K + k0 + koff;
    fa0.h[0] = *(const ushort8*)(ap0);
    fa0.h[1] = *(const ushort8*)(ap0 + 16);
    const unsigned short* ap1 = ap0 + (size_t)16 * K;
    fa1.h[0] = *(const ushort8*)(ap1);
    fa1.h[1] = *(const ushort8*)(ap1 + 16);
    #pragma unroll
    for (int j = 0; j < 4; ++j) {
      Frag fb;
      const unsigned short* bp = Bp + (((size_t)kt * NT + nt0 + j) * 32 + lane) * 16;
      fb.h[0] = *(const ushort8*)(bp);
      fb.h[1] = *(const ushort8*)(bp + 8);
      acc[0][j] = wmma_bf16(fa0.v, fb.v, acc[0][j]);
      acc[1][j] = wmma_bf16(fa1.v, fb.v, acc[1][j]);
    }
  }
  #pragma unroll
  for (int j = 0; j < 4; ++j) {
    int n = (nt0 + j) * 16 + (lane & 15);
    if (n >= N) continue;
    float bb = bias ? bias[n] : 0.f;
    #pragma unroll
    for (int i = 0; i < 2; ++i) {
      int rbase = m0 + i * 16 + ((lane >> 4) << 3);
      #pragma unroll
      for (int v = 0; v < 8; ++v) {
        float val = acc[i][j][v] + bb;
        float* cp = C + (size_t)(rbase + v) * N + n;
        if (addResidual) val += *cp;
        *cp = val;
      }
    }
  }
}

// ---------------- depthwise causal conv (K=4) + SiLU over xBC channels ----------------
__global__ void k_conv(const float* __restrict__ zx, const float* __restrict__ cw,
                       const float* __restrict__ cb, float* __restrict__ xconv) {
  int t = blockIdx.x * 256 + threadIdx.x;            // over BATCH*CONV_DIM
  if (t >= BATCH * CONV_DIM) return;
  int b = t / CONV_DIM, c = t - b * CONV_DIM;
  float w0 = cw[c*4+0], w1 = cw[c*4+1], w2 = cw[c*4+2], w3 = cw[c*4+3];
  float bias = cb[c];
  const float* src = zx + ((size_t)b * LSEQ) * D_IN_PROJ + D_INNER + c;
  float* dst = xconv + ((size_t)b * LSEQ) * CONV_DIM + c;
  float x0 = 0.f, x1 = 0.f, x2 = 0.f;
  for (int l = 0; l < LSEQ; ++l) {
    float x3 = src[(size_t)l * D_IN_PROJ];
    float a = bias + w0*x0 + w1*x1 + w2*x2 + w3*x3;
    dst[(size_t)l * CONV_DIM] = a * sigmoid_(a);
    x0 = x1; x1 = x2; x2 = x3;
  }
}

// ---------------- dt = softplus(dt_raw + dt_bias) ----------------
__global__ void k_dt(const float* __restrict__ zx, const float* __restrict__ dtb,
                     float* __restrict__ dt) {
  int t = blockIdx.x * 256 + threadIdx.x;            // over ROWS*NHEADS
  int r = t / NHEADS, h = t - r * NHEADS;
  float v = zx[(size_t)r * D_IN_PROJ + (D_INNER + CONV_DIM) + h] + dtb[h];
  dt[t] = (v > 20.f) ? v : log1pf(expf(v));
}

// ---------------- sequential SSD scan: one block per (batch, head) ----------------
// Software-pipelined: next step's B/C/x/dt global loads are issued right after the
// LDS-publish barrier so their latency hides under the current step's 64-FMA chain.
__global__ void __launch_bounds__(256)
k_scan(const float* __restrict__ xconv, const float* __restrict__ dtbuf,
       const float* __restrict__ A_log, const float* __restrict__ Dp,
       float* __restrict__ yscan) {
  int bh = blockIdx.x; int b = bh / NHEADS, h = bh - b * NHEADS;
  int t = threadIdx.x;                               // 256 threads
  int p = t >> 2;                                    // headdim element 0..63
  int nb = (t & 3) * 32;                             // 32 states per thread
  float A = -expf(A_log[h]);
  float D = Dp[h];
  float hs[32];
  #pragma unroll
  for (int n = 0; n < 32; ++n) hs[n] = 0.f;
  __shared__ float sB[D_STATE], sC[D_STATE];
  const float* base = xconv + (size_t)b * LSEQ * CONV_DIM;
  // per-thread staged-load offsets: threads 0..127 fetch B[t], 128..255 fetch C[t-128]
  int bcOff = (t < 128) ? (D_INNER + t) : (D_INNER + D_STATE + (t - 128));
  // prologue: fetch step 0
  float bcPre = base[bcOff];
  float xPre  = base[h * HEADDIM + p];
  float dtPre = dtbuf[(size_t)(b * LSEQ) * NHEADS + h];
  for (int l = 0; l < LSEQ; ++l) {
    // publish current step's B/C to LDS
    if (t < 128) sB[t] = bcPre; else sC[t - 128] = bcPre;
    float x = xPre, dt = dtPre;
    __syncthreads();
    // issue next step's loads (consumed at end of iteration -> latency hidden)
    float bcNext = 0.f, xNext = 0.f, dtNext = 0.f;
    if (l + 1 < LSEQ) {
      const float* rowp1 = base + (size_t)(l + 1) * CONV_DIM;
      bcNext = rowp1[bcOff];
      xNext  = rowp1[h * HEADDIM + p];
      dtNext = dtbuf[(size_t)(b * LSEQ + l + 1) * NHEADS + h];
    }
    float dA  = expf(dt * A);
    float dtx = dt * x;
    float part = 0.f;
    #pragma unroll
    for (int n = 0; n < 32; ++n) {
      hs[n] = hs[n] * dA + dtx * sB[nb + n];
      part += hs[n] * sC[nb + n];
    }
    part += __shfl_xor(part, 1, 32);
    part += __shfl_xor(part, 2, 32);
    if ((t & 3) == 0)
      yscan[(size_t)(b * LSEQ + l) * D_INNER + h * HEADDIM + p] = part + D * x;
    __syncthreads();
    bcPre = bcNext; xPre = xNext; dtPre = dtNext;
  }
}

// ---------------- gate with silu(z), RMSNorm, -> bf16 ----------------
__global__ void k_gate_rms(const float* __restrict__ yscan, const float* __restrict__ zx,
                           const float* __restrict__ rw, unsigned short* __restrict__ yn) {
  int r = blockIdx.x, t = threadIdx.x;               // 256 threads, 6 elems each
  __shared__ float red[256];
  float v[6]; float ss = 0.f;
  #pragma unroll
  for (int i = 0; i < 6; ++i) {
    int d = t + i * 256;
    float y = yscan[(size_t)r * D_INNER + d];
    float z = zx[(size_t)r * D_IN_PROJ + d];
    float g = y * (z * sigmoid_(z));
    v[i] = g; ss += g * g;
  }
  red[t] = ss; __syncthreads();
  for (int st = 128; st > 0; st >>= 1) { if (t < st) red[t] += red[t + st]; __syncthreads(); }
  float scale = rsqrtf(red[0] * (1.f / D_INNER) + 1e-5f);
  #pragma unroll
  for (int i = 0; i < 6; ++i) {
    int d = t + i * 256;
    yn[(size_t)r * D_INNER + d] = f2bf(v[i] * scale * rw[d]);
  }
}

extern "C" void kernel_launch(void* const* d_in, const int* in_sizes, int n_in,
                              void* d_out, int out_size, void* d_ws, size_t ws_size,
                              hipStream_t stream) {
  const int*   tokens  = (const int*)  d_in[0];
  const float* emb     = (const float*)d_in[1];
  const float* ln_w    = (const float*)d_in[2];
  const float* ln_b    = (const float*)d_in[3];
  const float* Win     = (const float*)d_in[4];
  const float* conv_w  = (const float*)d_in[5];
  const float* conv_b  = (const float*)d_in[6];
  const float* dt_bias = (const float*)d_in[7];
  const float* A_log   = (const float*)d_in[8];
  const float* Dp      = (const float*)d_in[9];
  const float* rms_w   = (const float*)d_in[10];
  const float* Wout    = (const float*)d_in[11];
  const float* Wcls    = (const float*)d_in[12];
  const float* bcls    = (const float*)d_in[13];

  char* p = (char*)d_ws;
  auto carve = [&](size_t bytes) -> char* {
    char* q = p; p += (bytes + 255) & ~(size_t)255; return q;
  };
  float*          x      = (float*)         carve((size_t)ROWS * D_MODEL * 4);
  unsigned short* nx     = (unsigned short*)carve((size_t)ROWS * D_MODEL * 2);
  float*          zx     = (float*)         carve((size_t)ROWS * D_IN_PROJ * 4);
  float*          xconv  = (float*)         carve((size_t)ROWS * CONV_DIM * 4);
  float*          dtb    = (float*)         carve((size_t)ROWS * NHEADS * 4);
  float*          yscan  = (float*)         carve((size_t)ROWS * D_INNER * 4);
  unsigned short* ynorm  = (unsigned short*)carve((size_t)ROWS * D_INNER * 2);
  unsigned short* xcls   = (unsigned short*)carve((size_t)ROWS * D_MODEL * 2);
  const int NT_IN = 212, NT_OUT = 48, NT_CLS = 64;   // padded N tile counts
  unsigned short* WinP   = (unsigned short*)carve((size_t)(D_MODEL/32) * NT_IN  * 512 * 2);
  unsigned short* WoutP  = (unsigned short*)carve((size_t)(D_INNER/32) * NT_OUT * 512 * 2);
  unsigned short* WclsP  = (unsigned short*)carve((size_t)(D_MODEL/32) * NT_CLS * 512 * 2);

  // embedding gather
  k_embed<<<(ROWS * D_MODEL) / 256, 256, 0, stream>>>(tokens, emb, x);

  for (int blk = 0; blk < 2; ++blk) {
    const float* Win_b  = Win  + (size_t)blk * D_MODEL * D_IN_PROJ;
    const float* Wout_b = Wout + (size_t)blk * D_INNER * D_MODEL;

    // pack in_proj weights, layernorm
    k_pack<<<(int)(((size_t)(D_MODEL/32) * NT_IN * 512) / 256), 256, 0, stream>>>(
        Win_b, WinP, D_MODEL, D_IN_PROJ, NT_IN);
    k_layernorm<<<ROWS, 256, 0, stream>>>(x, ln_w + blk * D_MODEL, ln_b + blk * D_MODEL, nx);

    // in_proj GEMM: (1024 x 768) @ (768 x 3352)
    {
      dim3 g(ROWS / 32, NT_IN / 4);
      k_gemm_bf16<<<g, 32, 0, stream>>>(nx, WinP, zx, nullptr, D_MODEL, D_IN_PROJ, NT_IN, 0);
    }

    // dt softplus + causal depthwise conv + silu
    k_dt<<<(ROWS * NHEADS) / 256, 256, 0, stream>>>(zx, dt_bias + blk * NHEADS, dtb);
    k_conv<<<(BATCH * CONV_DIM + 255) / 256, 256, 0, stream>>>(
        zx, conv_w + (size_t)blk * CONV_DIM * 4, conv_b + blk * CONV_DIM, xconv);

    // sequential SSD scan (state in registers, B/C staged in LDS, pipelined loads)
    k_scan<<<BATCH * NHEADS, 256, 0, stream>>>(
        xconv, dtb, A_log + blk * NHEADS, Dp + blk * NHEADS, yscan);

    // gate + rmsnorm -> bf16
    k_gate_rms<<<ROWS, 256, 0, stream>>>(yscan, zx, rms_w + blk * D_INNER, ynorm);

    // out_proj GEMM with residual add into x: (1024 x 1536) @ (1536 x 768)
    k_pack<<<(int)(((size_t)(D_INNER/32) * NT_OUT * 512) / 256), 256, 0, stream>>>(
        Wout_b, WoutP, D_INNER, D_MODEL, NT_OUT);
    {
      dim3 g(ROWS / 32, NT_OUT / 4);
      k_gemm_bf16<<<g, 32, 0, stream>>>(ynorm, WoutP, x, nullptr, D_INNER, D_MODEL, NT_OUT, 1);
    }
  }

  // classifier: (1024 x 768) @ (768 x 1000) + bias
  k_convert<<<(ROWS * D_MODEL) / 256, 256, 0, stream>>>(x, xcls);
  k_pack<<<(int)(((size_t)(D_MODEL/32) * NT_CLS * 512) / 256), 256, 0, stream>>>(
      Wcls, WclsP, D_MODEL, LABELS, NT_CLS);
  {
    dim3 g(ROWS / 32, NT_CLS / 4);
    k_gemm_bf16<<<g, 32, 0, stream>>>(xcls, WclsP, (float*)d_out, bcls, D_MODEL, LABELS, NT_CLS, 0);
  }
}
